// LlamaAttention_21053929685152
// MI455X (gfx1250) — matrix operand order repaired and checked by reading, not compiled, not run on hardware
//
#include <hip/hip_runtime.h>

// ---------------------------------------------------------------------------
// CDNA5 (gfx1250) Llama attention: bf16 WMMA everywhere, f32 accumulate.
// B=2 S=2048 H=4096 NH=32 NKV=8 HD=128, wave32.
// ---------------------------------------------------------------------------

typedef __bf16 bf16_t;
typedef __bf16 bf16x8_t  __attribute__((ext_vector_type(8)));
typedef __bf16 bf16x16_t __attribute__((ext_vector_type(16)));
typedef float  f32x8_t   __attribute__((ext_vector_type(8)));

#define DEV static __device__ __forceinline__

constexpr int B_   = 2;
constexpr int S_   = 2048;
constexpr int H_   = 4096;
constexpr int NH_  = 32;
constexpr int NKV_ = 8;
constexpr int HD_  = 128;
constexpr int M_   = B_ * S_;            // 4096 token rows
constexpr float SCALE_ = 0.08838834764831845f;  // HD^-0.5

DEV bf16x8_t ld8(const bf16_t* p) { return *reinterpret_cast<const bf16x8_t*>(p); }

DEV bf16x16_t join16(bf16x8_t lo, bf16x8_t hi) {
  return __builtin_shufflevector(lo, hi, 0,1,2,3,4,5,6,7,8,9,10,11,12,13,14,15);
}

DEV f32x8_t zero8() { f32x8_t z = {0.f,0.f,0.f,0.f,0.f,0.f,0.f,0.f}; return z; }

DEV f32x8_t wmma_bf16(bf16x16_t a, bf16x16_t b, f32x8_t c) {
  // D = A(16x32 bf16) * B(32x16 bf16) + C(16x16 f32)
  return __builtin_amdgcn_wmma_f32_16x16x32_bf16(
      /*neg_a=*/false, a, /*neg_b=*/false, b,
      /*c_mod=*/(short)0, c, /*reuse_a=*/false, /*reuse_b=*/false);
}

// ---------------------------------------------------------------------------
// fp32 -> bf16 bulk convert (8 elems / thread, b128 in / b128 out)
// ---------------------------------------------------------------------------
__global__ void __launch_bounds__(256)
k_cvt(const float* __restrict__ in, bf16_t* __restrict__ out, int n8)
{
  int i = blockIdx.x * 256 + threadIdx.x;
  if (i >= n8) return;
  const float4* p = reinterpret_cast<const float4*>(in) + (size_t)i * 2;
  float4 a = p[0], b4 = p[1];
  bf16x8_t v;
  v[0] = (bf16_t)a.x;  v[1] = (bf16_t)a.y;  v[2] = (bf16_t)a.z;  v[3] = (bf16_t)a.w;
  v[4] = (bf16_t)b4.x; v[5] = (bf16_t)b4.y; v[6] = (bf16_t)b4.z; v[7] = (bf16_t)b4.w;
  reinterpret_cast<bf16x8_t*>(out)[i] = v;
}

// ---------------------------------------------------------------------------
// QKV projection GEMM + RoPE + transposed store.
// C[M, nheads*128] = A[M,4096] * W[nheads*128, 4096]^T  (NT: K contiguous both)
// One wave = 16 rows x 128 cols (one full head) -> RoPE partner d±64 is tile
// t^4 of the same wave. Output layout: [B, nheads, S, HD] bf16.
// ---------------------------------------------------------------------------
__global__ void __launch_bounds__(128)
k_gemm_qkv(const bf16_t* __restrict__ A, const bf16_t* __restrict__ W,
           bf16_t* __restrict__ outT, const float* __restrict__ cosp,
           const float* __restrict__ sinp, int nheads, int doRope)
{
  const int ln   = threadIdx.x & 31;
  const int gw   = blockIdx.x * 4 + (threadIdx.x >> 5);
  const int mt   = gw % (M_ / 16);
  const int head = gw / (M_ / 16);
  const int am   = ln & 15;               // M (A) / N (B) index within tile
  const int koff = (ln < 16) ? 0 : 8;     // unused; A loads its K half at kb, like B
  const int kb   = (ln < 16) ? 0 : 16;    // B-fragment K half

  const bf16_t* Ap = A + (size_t)(mt * 16 + am) * H_;
  const bf16_t* Wp = W + (size_t)head * HD_ * H_;

  f32x8_t acc[8];
#pragma unroll
  for (int t = 0; t < 8; ++t) acc[t] = zero8();

  for (int k0 = 0; k0 < H_; k0 += 32) {
    __builtin_prefetch(Ap + k0 + 512, 0, 0);   // global_prefetch_b8
    bf16x16_t af = join16(ld8(Ap + k0 + kb), ld8(Ap + k0 + kb + 8));
#pragma unroll
    for (int t = 0; t < 8; ++t) {
      const bf16_t* wp = Wp + (size_t)(t * 16 + am) * H_ + k0 + kb;
      bf16x16_t bfr = join16(ld8(wp), ld8(wp + 8));
      acc[t] = wmma_bf16(af, bfr, acc[t]);
    }
  }

  const int rbase = (ln < 16) ? 0 : 8;
#pragma unroll
  for (int r = 0; r < 8; ++r) {
    const int mrow = mt * 16 + r + rbase;
    const int b = mrow >> 11;             // / S_
    const int s = mrow & (S_ - 1);
    const size_t obase = ((size_t)(b * nheads + head) * S_ + s) * HD_;
    if (doRope) {
      const size_t cbase = ((size_t)b * S_ + s) * HD_;
#pragma unroll
      for (int t = 0; t < 8; ++t) {
        const int d = t * 16 + am;
        float c  = cosp[cbase + d];
        float sn = sinp[cbase + d];
        float x  = acc[t][r];
        float xr = (t < 4) ? -acc[t + 4][r] : acc[t - 4][r];  // rotate_half
        outT[obase + d] = (bf16_t)(c * x + sn * xr);
      }
    } else {
#pragma unroll
      for (int t = 0; t < 8; ++t)
        outT[obase + t * 16 + am] = (bf16_t)acc[t][r];
    }
  }
}

// ---------------------------------------------------------------------------
// Flash attention: one wave per 16-query tile of one head; 32-key chunks.
// Per chunk: 8 WMMA (QK^T, K-dim=HD=128) + online softmax + 8 WMMA (P*V).
// P transposed C-layout -> A-layout via per-wave LDS; V staged in LDS.
// ---------------------------------------------------------------------------
constexpr int PSTR = 40;                     // p-buffer row stride (bf16)
constexpr int VSTR = 136;                    // v-buffer row stride (bf16), 272B (16B mult)
constexpr int WLDS = 16 * PSTR + 32 * VSTR;  // 4992 bf16 per wave

__global__ void __launch_bounds__(128)
k_attn(const bf16_t* __restrict__ Q, const bf16_t* __restrict__ K,
       const bf16_t* __restrict__ V, bf16_t* __restrict__ ctx)
{
  __shared__ __align__(16) bf16_t sh[4 * WLDS];
  const int ln  = threadIdx.x & 31;
  const int wv  = threadIdx.x >> 5;
  const int gw  = blockIdx.x * 4 + wv;
  const int qt  = gw & (S_ / 16 - 1);        // query tile
  const int h   = (gw >> 7) & (NH_ - 1);
  const int b   = gw >> 12;
  const int kvh = h >> 2;                    // GQA: NH/NKV = 4

  bf16_t* pbuf = sh + wv * WLDS;
  bf16_t* vbuf = pbuf + 16 * PSTR;

  const bf16_t* Qp = Q + (((size_t)(b * NH_ + h)) * S_ + qt * 16) * HD_;
  const bf16_t* Kp = K + ((size_t)(b * NKV_ + kvh)) * S_ * HD_;
  const bf16_t* Vp = V + ((size_t)(b * NKV_ + kvh)) * S_ * HD_;

  const int am    = ln & 15;
  const int koff  = (ln < 16) ? 0 : 8;
  const int kb    = (ln < 16) ? 0 : 16;
  const int rbase = (ln < 16) ? 0 : 8;

  // Q fragments, resident for the whole K loop (16x128 bf16)
  bf16x16_t qf[4];
#pragma unroll
  for (int ks = 0; ks < 4; ++ks) {
    const bf16_t* qp = Qp + (size_t)am * HD_ + ks * 32;
    qf[ks] = join16(ld8(qp + kb), ld8(qp + kb + 8));
  }

  f32x8_t acc[8];
#pragma unroll
  for (int t = 0; t < 8; ++t) acc[t] = zero8();
  float rm[8], rl[8];
#pragma unroll
  for (int r = 0; r < 8; ++r) { rm[r] = -3.0e38f; rl[r] = 0.f; }

  const int nchunk = (qt * 16 + 16 + 31) >> 5;   // causal: keys 0..qt*16+15
  for (int c = 0; c < nchunk; ++c) {
    const int k0 = c * 32;

    // Stage 32x128 V chunk into LDS with coalesced b128 loads.
#pragma unroll
    for (int i = 0; i < 16; ++i) {
      const int idx = i * 32 + ln;               // 0..511
      const int row = idx >> 4;
      const int cc  = idx & 15;
      bf16x8_t vv = ld8(Vp + (size_t)(k0 + row) * HD_ + cc * 8);
      *reinterpret_cast<bf16x8_t*>(vbuf + row * VSTR + cc * 8) = vv;
    }

    // S = Q * K^T  (two 16x16 score tiles, K-dim = 128)
    f32x8_t sc0 = zero8(), sc1 = zero8();
#pragma unroll
    for (int ks = 0; ks < 4; ++ks) {
      const bf16_t* kp0 = Kp + (size_t)(k0 + am) * HD_ + ks * 32 + kb;
      const bf16_t* kp1 = Kp + (size_t)(k0 + 16 + am) * HD_ + ks * 32 + kb;
      sc0 = wmma_bf16(qf[ks], join16(ld8(kp0), ld8(kp0 + 8)), sc0);
      sc1 = wmma_bf16(qf[ks], join16(ld8(kp1), ld8(kp1 + 8)), sc1);
    }

    // scale + causal mask + online softmax (row lives in one VGPR / lane-half)
    const int key0 = k0 + am;
    const int key1 = k0 + 16 + am;
#pragma unroll
    for (int r = 0; r < 8; ++r) {
      const int qrow = qt * 16 + r + rbase;
      float s0 = sc0[r] * SCALE_;
      float s1 = sc1[r] * SCALE_;
      if (key0 > qrow) s0 = -1.0e30f;
      if (key1 > qrow) s1 = -1.0e30f;
      float cm = fmaxf(s0, s1);
#pragma unroll
      for (int o = 1; o < 16; o <<= 1) cm = fmaxf(cm, __shfl_xor(cm, o, 32));
      float mnew  = fmaxf(rm[r], cm);
      float alpha = __expf(rm[r] - mnew);
      float p0 = __expf(s0 - mnew);
      float p1 = __expf(s1 - mnew);
      sc0[r] = p0; sc1[r] = p1;
      float ps = p0 + p1;
#pragma unroll
      for (int o = 1; o < 16; o <<= 1) ps += __shfl_xor(ps, o, 32);
      rl[r] = rl[r] * alpha + ps;
      rm[r] = mnew;
#pragma unroll
      for (int t = 0; t < 8; ++t) acc[t][r] *= alpha;
    }

    // P: C-layout -> A-layout transpose through LDS
#pragma unroll
    for (int r = 0; r < 8; ++r) {
      pbuf[(r + rbase) * PSTR + am]      = (bf16_t)sc0[r];
      pbuf[(r + rbase) * PSTR + 16 + am] = (bf16_t)sc1[r];
    }
    asm volatile("s_wait_dscnt 0x0" ::: "memory");
    bf16x16_t pf;
#pragma unroll
    for (int j = 0; j < 16; ++j) {
      const int kk = kb + j;
      pf[j] = pbuf[am * PSTR + kk];
    }

    // O += P(16x32) * V(32x128)
#pragma unroll
    for (int t = 0; t < 8; ++t) {
      bf16x16_t vf;
      const int d = t * 16 + am;
#pragma unroll
      for (int j = 0; j < 16; ++j)
        vf[j] = vbuf[(kb + j) * VSTR + d];
      acc[t] = wmma_bf16(pf, vf, acc[t]);
    }
  }

  // finalize: divide by row sum, store ctx [B, S, NH*HD] bf16
#pragma unroll
  for (int r = 0; r < 8; ++r) {
    const float inv = 1.0f / rl[r];
    const int s = qt * 16 + r + rbase;
    const size_t obase = ((size_t)b * S_ + s) * (NH_ * HD_) + (size_t)h * HD_;
#pragma unroll
    for (int t = 0; t < 8; ++t)
      ctx[obase + t * 16 + am] = (bf16_t)(acc[t][r] * inv);
  }
}

// ---------------------------------------------------------------------------
// Output projection: out[M,4096] f32 = ctx[M,4096] bf16 * Wo[4096,4096]^T
// ---------------------------------------------------------------------------
__global__ void __launch_bounds__(128)
k_gemm_out(const bf16_t* __restrict__ A, const bf16_t* __restrict__ W,
           float* __restrict__ out)
{
  const int ln = threadIdx.x & 31;
  const int gw = blockIdx.x * 4 + (threadIdx.x >> 5);
  const int mt = gw % (M_ / 16);
  const int nb = gw / (M_ / 16);            // 128-wide column block
  const int am   = ln & 15;
  const int koff = (ln < 16) ? 0 : 8;
  const int kb   = (ln < 16) ? 0 : 16;

  const bf16_t* Ap = A + (size_t)(mt * 16 + am) * H_;
  const bf16_t* Wp = W + (size_t)nb * HD_ * H_;

  f32x8_t acc[8];
#pragma unroll
  for (int t = 0; t < 8; ++t) acc[t] = zero8();

  for (int k0 = 0; k0 < H_; k0 += 32) {
    __builtin_prefetch(Ap + k0 + 512, 0, 0);
    bf16x16_t af = join16(ld8(Ap + k0 + kb), ld8(Ap + k0 + kb + 8));
#pragma unroll
    for (int t = 0; t < 8; ++t) {
      const bf16_t* wp = Wp + (size_t)(t * 16 + am) * H_ + k0 + kb;
      acc[t] = wmma_bf16(af, join16(ld8(wp), ld8(wp + 8)), acc[t]);
    }
  }

  const int rbase = (ln < 16) ? 0 : 8;
#pragma unroll
  for (int r = 0; r < 8; ++r) {
    const int mrow = mt * 16 + r + rbase;
#pragma unroll
    for (int t = 0; t < 8; ++t)
      out[(size_t)mrow * H_ + (size_t)nb * HD_ + t * 16 + am] = acc[t][r];
  }
}

// ---------------------------------------------------------------------------
// Launch: convert -> QKV(+RoPE, transposed) -> flash attention -> O-proj.
// Workspace layout (~201 MB, 256B-aligned slabs).
// ---------------------------------------------------------------------------
extern "C" void kernel_launch(void* const* d_in, const int* in_sizes, int n_in,
                              void* d_out, int out_size, void* d_ws, size_t ws_size,
                              hipStream_t stream)
{
  (void)in_sizes; (void)n_in; (void)out_size; (void)ws_size;
  const float* hs   = (const float*)d_in[0];
  const float* cosp = (const float*)d_in[1];
  const float* sinp = (const float*)d_in[2];
  // d_in[3] = attention_mask (pure causal; recomputed analytically)
  const float* Wq   = (const float*)d_in[4];
  const float* Wk   = (const float*)d_in[5];
  const float* Wv   = (const float*)d_in[6];
  const float* Wo   = (const float*)d_in[7];

  char* ws = (char*)d_ws;
  size_t off = 0;
  auto take = [&](size_t elems) -> bf16_t* {
    bf16_t* p = (bf16_t*)(ws + off);
    off += ((elems * sizeof(bf16_t)) + 255) & ~(size_t)255;
    return p;
  };
  bf16_t* hsb = take((size_t)M_ * H_);            // 33.5 MB
  bf16_t* wqb = take((size_t)NH_ * HD_ * H_);     // 33.5 MB
  bf16_t* wkb = take((size_t)NKV_ * HD_ * H_);    //  8.4 MB
  bf16_t* wvb = take((size_t)NKV_ * HD_ * H_);    //  8.4 MB
  bf16_t* wob = take((size_t)H_ * H_);            // 33.5 MB
  bf16_t* qT  = take((size_t)B_ * NH_ * S_ * HD_);  // 33.5 MB
  bf16_t* kT  = take((size_t)B_ * NKV_ * S_ * HD_); //  8.4 MB
  bf16_t* vT  = take((size_t)B_ * NKV_ * S_ * HD_); //  8.4 MB
  bf16_t* ctx = take((size_t)M_ * NH_ * HD_);       // 33.5 MB

  auto cvt = [&](const float* in, bf16_t* out, size_t n) {
    int n8 = (int)(n / 8);
    k_cvt<<<dim3((n8 + 255) / 256), dim3(256), 0, stream>>>(in, out, n8);
  };
  cvt(hs, hsb, (size_t)M_ * H_);
  cvt(Wq, wqb, (size_t)NH_ * HD_ * H_);
  cvt(Wk, wkb, (size_t)NKV_ * HD_ * H_);
  cvt(Wv, wvb, (size_t)NKV_ * HD_ * H_);
  cvt(Wo, wob, (size_t)H_ * H_);

  // Q/K with RoPE, V plain; outputs transposed to [B, heads, S, HD] bf16
  k_gemm_qkv<<<dim3((M_ / 16) * NH_ / 4),  dim3(128), 0, stream>>>(hsb, wqb, qT, cosp, sinp, NH_, 1);
  k_gemm_qkv<<<dim3((M_ / 16) * NKV_ / 4), dim3(128), 0, stream>>>(hsb, wkb, kT, cosp, sinp, NKV_, 1);
  k_gemm_qkv<<<dim3((M_ / 16) * NKV_ / 4), dim3(128), 0, stream>>>(hsb, wvb, vT, nullptr, nullptr, NKV_, 0);

  k_attn<<<dim3(B_ * NH_ * (S_ / 16) / 4), dim3(128), 0, stream>>>(qT, kT, vT, ctx);

  k_gemm_out<<<dim3((M_ / 16) * (H_ / HD_) / 4), dim3(128), 0, stream>>>(ctx, wob, (float*)d_out);
}
